// GMCR2Layer_85177791414430
// MI455X (gfx1250) — compile-verified, matrix operand order checked by reading, not run on hardware
//
#include <hip/hip_runtime.h>

#define NN 4096
#define DD 256
#define SS 32
#define KK 4
#define ETA 0.5f

typedef __attribute__((ext_vector_type(16))) _Float16 v16h;
typedef __attribute__((ext_vector_type(8)))  _Float16 v8h;
typedef __attribute__((ext_vector_type(8)))  float    v8f;

// Build a v16h A/B operand from two contiguous 16-byte chunks.
// CDNA5 16-bit A/B layout (ISA 7.12.2): lanes 0-15 hold K-chunks {0..7,16..23},
// lanes 16-31 hold {8..15,24..31}; v16h element e -> VGPR e/2, half e%2.
__device__ __forceinline__ v16h make_ab(const _Float16* p0, const _Float16* p1) {
  v8h lo = *(const v8h*)p0;
  v8h hi = *(const v8h*)p1;
  return __builtin_shufflevector(lo, hi, 0,1,2,3,4,5,6,7,8,9,10,11,12,13,14,15);
}

// ---------------------------------------------------------------------------
// Kernel 1: Z = H @ U_k  -> f16 operand copies (scaled-query, key, value^T)
// ---------------------------------------------------------------------------
__global__ __launch_bounds__(32) void zproj_kernel(
    const float* __restrict__ H, const float* __restrict__ U,
    const float* __restrict__ lp,
    _Float16* __restrict__ Zs, _Float16* __restrict__ Zf, _Float16* __restrict__ ZT)
{
  const int n = blockIdx.x;
  const int k = blockIdx.y;
  const int s = threadIdx.x;                      // 0..31
  const float* Hrow = H + (size_t)n * DD;
  const float* Ucol = U + (size_t)k * DD * SS + s;
  float acc = 0.f;
  #pragma unroll 4
  for (int d = 0; d < DD; ++d) acc = fmaf(Hrow[d], Ucol[(size_t)d * SS], acc);
  const float e = __expf(lp[k * SS + s]);
  Zf[((size_t)k * NN + n) * SS + s] = (_Float16)acc;            // keys (score B)
  Zs[((size_t)k * NN + n) * SS + s] = (_Float16)(acc * e);      // queries (score A)
  ZT[((size_t)k * SS + s) * NN + n] = (_Float16)acc;            // values (PV B)
}

// ---------------------------------------------------------------------------
// Kernel 2: fused masked attention (flash style), one wave per 16-query tile,
// 64-key blocks. p = mask * exp(score - rowmax); rowsum via ones-column WMMA.
// ---------------------------------------------------------------------------
__global__ __launch_bounds__(32) void flash_attn_kernel(
    const _Float16* __restrict__ Zs, const _Float16* __restrict__ Zf,
    const _Float16* __restrict__ ZT, const float* __restrict__ adj,
    float* __restrict__ O)
{
  const int k    = blockIdx.y;
  const int qb   = blockIdx.x * 16;
  const int l    = threadIdx.x;        // 0..31
  const int ll   = l & 15;
  const int half = l >> 4;
  const int c0   = half ? 8 : 0;       // per-half K-chunk base (A/B layout)

  __shared__ _Float16 __attribute__((aligned(16))) Plds[16 * 64];

  const float inv_s = 0.17677669529663687f;   // 1/sqrt(32)

  // A operand: precision-scaled query tile, row = lane%16 (loaded once)
  const _Float16* qrow = Zs + ((size_t)k * NN + (qb + ll)) * SS;
  const v16h aQ = make_ab(qrow + c0, qrow + c0 + 16);

  v16h ones;
  #pragma unroll
  for (int i = 0; i < 16; ++i) ones[i] = (_Float16)1.0f;

  float m[8];
  #pragma unroll
  for (int v = 0; v < 8; ++v) m[v] = -INFINITY;
  v8f o0 = {}; v8f o1 = {};
  v8f lacc = {};                        // softmax denominator tile (P @ ones)

  for (int kb = 0; kb < NN; kb += 64) {
    // ---- scores: 4 key tiles of 16 (K = s = 32, one WMMA each) ----
    v8f sc[4];
    #pragma unroll
    for (int t = 0; t < 4; ++t) {
      const _Float16* kr = Zf + ((size_t)k * NN + (kb + t * 16 + ll)) * SS;
      const v16h bK = make_ab(kr + c0, kr + c0 + 16);
      v8f z = {};
      sc[t] = __builtin_amdgcn_wmma_f32_16x16x32_f16(false, aQ, false, bK, (short)0, z, false, false);
    }

    // ---- online softmax over this 64-key block ----
    // C layout: lane -> key column (mod 16), VGPR v -> query row v + 8*half
    #pragma unroll
    for (int v = 0; v < 8; ++v) {
      const int q = qb + v + 8 * half;
      const float* mrow = adj + (size_t)q * NN + kb + ll;
      const float w0 = mrow[0], w1 = mrow[16], w2 = mrow[32], w3 = mrow[48];
      __builtin_prefetch(mrow + 64, 0, 0);          // next block's mask row
      const float r0 = sc[0][v] * inv_s;
      const float r1 = sc[1][v] * inv_s;
      const float r2 = sc[2][v] * inv_s;
      const float r3 = sc[3][v] * inv_s;
      float mx = fmaxf(fmaxf(r0, r1), fmaxf(r2, r3));
      #pragma unroll
      for (int off = 1; off < 16; off <<= 1) mx = fmaxf(mx, __shfl_xor(mx, off, 32));
      const float mn    = fmaxf(m[v], mx);
      const float scale = __expf(m[v] - mn);        // first iter: exp(-inf)=0
      m[v] = mn;
      const float p0 = w0 * __expf(r0 - mn);
      const float p1 = w1 * __expf(r1 - mn);
      const float p2 = w2 * __expf(r2 - mn);
      const float p3 = w3 * __expf(r3 - mn);
      o0[v]   *= scale;
      o1[v]   *= scale;
      lacc[v] *= scale;
      const int r = v + 8 * half;
      Plds[r * 64 + ll]      = (_Float16)p0;        // transpose P through LDS
      Plds[r * 64 + 16 + ll] = (_Float16)p1;
      Plds[r * 64 + 32 + ll] = (_Float16)p2;
      Plds[r * 64 + 48 + ll] = (_Float16)p3;
    }
    asm volatile("s_wait_dscnt 0" ::: "memory");    // LDS in-order within wave

    // ---- O += P @ Z ; lacc += P @ 1  (K = 64 keys -> two A chunks) ----
    #pragma unroll
    for (int c = 0; c < 2; ++c) {
      const v16h aP = make_ab(&Plds[ll * 64 + c * 32 + c0],
                              &Plds[ll * 64 + c * 32 + c0 + 16]);
      const _Float16* vr0 = ZT + ((size_t)k * SS + ll)      * NN + kb + c * 32;
      const _Float16* vr1 = ZT + ((size_t)k * SS + 16 + ll) * NN + kb + c * 32;
      const v16h bV0 = make_ab(vr0 + c0, vr0 + c0 + 16);
      const v16h bV1 = make_ab(vr1 + c0, vr1 + c0 + 16);
      o0   = __builtin_amdgcn_wmma_f32_16x16x32_f16(false, aP, false, bV0,  (short)0, o0,   false, false);
      o1   = __builtin_amdgcn_wmma_f32_16x16x32_f16(false, aP, false, bV1,  (short)0, o1,   false, false);
      lacc = __builtin_amdgcn_wmma_f32_16x16x32_f16(false, aP, false, ones, (short)0, lacc, false, false);
    }
  }

  #pragma unroll
  for (int v = 0; v < 8; ++v) {
    const float inv_l = 1.0f / lacc[v];             // all columns of lacc equal rowsum
    const int q = qb + v + 8 * half;
    O[((size_t)k * NN + q) * SS + ll]      = o0[v] * inv_l;
    O[((size_t)k * NN + q) * SS + 16 + ll] = o1[v] * inv_l;
  }
}

// ---------------------------------------------------------------------------
// Kernel 3: agg = sum_k O_k U_k^T ; H_half = H + eta*agg ; soft threshold
// ---------------------------------------------------------------------------
__global__ __launch_bounds__(256) void combine_kernel(
    const float* __restrict__ H, const float* __restrict__ U,
    const float* __restrict__ O, const float* __restrict__ thr,
    float* __restrict__ out)
{
  const int n = blockIdx.x;
  const int d = threadIdx.x;
  __shared__ float Os[KK * SS];
  if (d < KK * SS) Os[d] = O[((size_t)(d >> 5) * NN + n) * SS + (d & 31)];
  __syncthreads();
  float agg = 0.f;
  #pragma unroll
  for (int k = 0; k < KK; ++k) {
    const float* Urow = U + ((size_t)k * DD + d) * SS;
    #pragma unroll
    for (int s = 0; s < SS; ++s) agg = fmaf(Os[k * SS + s], Urow[s], agg);
  }
  const float h = H[(size_t)n * DD + d] + ETA * agg;
  const float a = fabsf(h) - thr[d];
  out[(size_t)n * DD + d] = (a > 0.f) ? copysignf(a, h) : 0.f;
}

// ---------------------------------------------------------------------------
// Kernel 4: orthogonality loss (tiny), single workgroup
// ---------------------------------------------------------------------------
__global__ __launch_bounds__(256) void orth_kernel(const float* __restrict__ U,
                                                   float* __restrict__ out)
{
  const int t = threadIdx.x;
  float acc = 0.f;
  for (int idx = t; idx < KK * KK * SS * SS; idx += 256) {
    const int b  = idx & (SS - 1);
    const int a  = (idx >> 5) & (SS - 1);
    const int lq = (idx >> 10) & (KK - 1);
    const int kq = idx >> 12;
    if (kq > lq) continue;
    float c = 0.f;
    const float* Ua = U + (size_t)kq * DD * SS + a;
    const float* Ub = U + (size_t)lq * DD * SS + b;
    for (int d = 0; d < DD; ++d) c = fmaf(Ua[(size_t)d * SS], Ub[(size_t)d * SS], c);
    if (kq == lq) { const float e = c - (a == b ? 1.f : 0.f); acc += e * e; }
    else          { acc += c * c; }
  }
  __shared__ float red[256];
  red[t] = acc; __syncthreads();
  for (int off = 128; off > 0; off >>= 1) {
    if (t < off) red[t] += red[t + off];
    __syncthreads();
  }
  if (t == 0) out[(size_t)NN * DD] = red[0];
}

// ---------------------------------------------------------------------------
extern "C" void kernel_launch(void* const* d_in, const int* in_sizes, int n_in,
                              void* d_out, int out_size, void* d_ws, size_t ws_size,
                              hipStream_t stream) {
  (void)in_sizes; (void)n_in; (void)out_size; (void)ws_size;
  const float* H   = (const float*)d_in[0];
  const float* adj = (const float*)d_in[1];
  const float* U   = (const float*)d_in[2];
  const float* lp  = (const float*)d_in[3];
  const float* thr = (const float*)d_in[4];
  float* out = (float*)d_out;

  // workspace: Zs(1MB) | Zf(1MB) | ZT(1MB) | O(2MB)  -> needs 5 MB
  char* ws = (char*)d_ws;
  _Float16* Zs = (_Float16*)(ws);
  _Float16* Zf = (_Float16*)(ws + (1u << 20));
  _Float16* ZT = (_Float16*)(ws + (2u << 20));
  float*    O  = (float*)   (ws + (3u << 20));

  zproj_kernel     <<<dim3(NN, KK),      32, 0, stream>>>(H, U, lp, Zs, Zf, ZT);
  flash_attn_kernel<<<dim3(NN / 16, KK), 32, 0, stream>>>(Zs, Zf, ZT, adj, O);
  combine_kernel   <<<NN,               DD, 0, stream>>>(H, U, O, thr, out);
  orth_kernel      <<<1,               256, 0, stream>>>(U, out);
}
